// GINEncoder_70849780515148
// MI455X (gfx1250) — matrix operand order repaired and checked by reading, not compiled, not run on hardware
//
#include <hip/hip_runtime.h>
#include <cstdint>
#include <cstddef>

// ---------------------------------------------------------------------------
// GIN encoder for MI455X (gfx1250, wave32).
// GEMMs: v_wmma_f32_16x16x32_bf16, 8 waves/block, block tile 256x64,
// wave tile 32x64 (8 accumulators), double-buffered LDS staging.
// Aggregation: float4 gather + global atomic f32 scatter-add (roofline-
// dominant phase, ~5 GB HBM traffic total).
// ---------------------------------------------------------------------------

typedef __bf16 bf16;
typedef bf16  v16bf __attribute__((ext_vector_type(16)));
typedef bf16  v8bf  __attribute__((ext_vector_type(8)));
typedef float v8f   __attribute__((ext_vector_type(8)));

#define HDIM 256
#define BN_EPS 1e-5f

__device__ __forceinline__ unsigned short f2bf(float f) {
  unsigned int u = __float_as_uint(f);
  unsigned int r = 0x7FFFu + ((u >> 16) & 1u);   // round-to-nearest-even
  u += r;
  return (unsigned short)(u >> 16);
}

// ---------------------------------------------------------------- zero fill
__global__ void k_zero(float* __restrict__ p, int n) {
  int i = blockIdx.x * blockDim.x + threadIdx.x;
  int stride = gridDim.x * blockDim.x;
  for (; i < n; i += stride) p[i] = 0.0f;
}

// -------------------------------------------------- edge scatter-add (agg)
__global__ void k_aggregate(const float* __restrict__ h,
                            const int* __restrict__ rows,
                            const int* __restrict__ cols,
                            float* __restrict__ agg,
                            int E, int D, int logD4) {
  int tid = blockIdx.x * blockDim.x + threadIdx.x;
  int total = E << logD4;
  if (tid >= total) return;
  int e = tid >> logD4;
  int c = (tid & ((1 << logD4) - 1)) << 2;
  int src = cols[e];
  int dst = rows[e];
  const float4 v = *(const float4*)(h + (size_t)src * D + c);
  float* p = agg + (size_t)dst * D + c;
  atomicAdd(p + 0, v.x);
  atomicAdd(p + 1, v.y);
  atomicAdd(p + 2, v.z);
  atomicAdd(p + 3, v.w);
}

// ------------------------------------- (1+eps)*h + agg  -> bf16 GEMM input
__global__ void k_combine(const float* __restrict__ h,
                          const float* __restrict__ agg,
                          const float* __restrict__ epsArr, int layer,
                          unsigned short* __restrict__ outbf, int n) {
  float s = 1.0f + epsArr[layer];
  int i = blockIdx.x * blockDim.x + threadIdx.x;
  int stride = gridDim.x * blockDim.x;
  for (; i < n; i += stride) outbf[i] = f2bf(s * h[i] + agg[i]);
}

// ----------------------------------------------------- f32 -> bf16 weights
__global__ void k_f2bf(const float* __restrict__ in,
                       unsigned short* __restrict__ out, int n) {
  int i = blockIdx.x * blockDim.x + threadIdx.x;
  int stride = gridDim.x * blockDim.x;
  for (; i < n; i += stride) out[i] = f2bf(in[i]);
}

// ---------------------------------------------------------------- WMMA GEMM
// C[M x 256] = A[M x K](bf16) * B[K x 256](bf16) + bias, C f32.
// Block: 256 threads (8 waves), tile 256(M) x 64(N), K stepped by 32.
// Wave tile 32x64: 2 A frags x 4 B frags -> 8 wmma per K-step.
// Double-buffered LDS: global loads of tile kt+1 overlap wmma of tile kt.
__global__ __launch_bounds__(256)
void k_gemm(const unsigned short* __restrict__ A,
            const unsigned short* __restrict__ B,
            const float* __restrict__ bias,
            float* __restrict__ C, int M, int K) {
  const int LDA = 40;                         // padded LDS stride (conflict-free)
  __shared__ unsigned short As[2][256 * 40];  // A tiles [row][k]   (256 x 32)
  __shared__ unsigned short Bs[2][64 * 40];   // B tiles [n][k] transposed (64 x 32)

  const int tid  = threadIdx.x;
  const int wave = tid >> 5;
  const int lane = tid & 31;
  const int hi   = lane >> 4;                 // lane half: 0 | 1
  const int l15  = lane & 15;
  const int rowBase = blockIdx.x * 256;
  const int colBase = blockIdx.y * 64;

  v8f acc[2][4];
#pragma unroll
  for (int mt = 0; mt < 2; ++mt)
#pragma unroll
    for (int t = 0; t < 4; ++t)
#pragma unroll
      for (int j = 0; j < 8; ++j) acc[mt][t][j] = 0.0f;

  const int KT = K >> 5;                      // K-steps of 32
  uint4 aReg[4];
  uint4 bReg;

  // ---- prologue: global-load tile 0 and stage into buffer 0
#pragma unroll
  for (int i = 0; i < 4; ++i) {
    int chunk = tid + (i << 8);               // 0..1023 (256 rows x 4 chunks)
    int r  = chunk >> 2;
    int cc = (chunk & 3) << 3;
    int gr = rowBase + r;
    aReg[i] = make_uint4(0u, 0u, 0u, 0u);
    if (gr < M) aReg[i] = *(const uint4*)(A + (size_t)gr * K + cc);
  }
  {
    int k  = tid >> 3;                        // 0..31
    int ns = (tid & 7) << 3;                  // 0..56
    bReg = *(const uint4*)(B + (size_t)k * HDIM + colBase + ns);
  }
#pragma unroll
  for (int i = 0; i < 4; ++i) {
    int chunk = tid + (i << 8);
    int r  = chunk >> 2;
    int cc = (chunk & 3) << 3;
    *(uint4*)(&As[0][r * LDA + cc]) = aReg[i];
  }
  {
    int k  = tid >> 3;
    int ns = (tid & 7) << 3;
    const unsigned short* pv = (const unsigned short*)&bReg;
#pragma unroll
    for (int j = 0; j < 8; ++j) Bs[0][(ns + j) * LDA + k] = pv[j];
  }
  __syncthreads();

  for (int kt = 0; kt < KT; ++kt) {
    const int cur = kt & 1;
    const bool more = (kt + 1) < KT;

    // ---- issue global loads for next tile (overlaps with wmma below)
    if (more) {
      const int k0 = (kt + 1) << 5;
#pragma unroll
      for (int i = 0; i < 4; ++i) {
        int chunk = tid + (i << 8);
        int r  = chunk >> 2;
        int cc = (chunk & 3) << 3;
        int gr = rowBase + r;
        aReg[i] = make_uint4(0u, 0u, 0u, 0u);
        if (gr < M) aReg[i] = *(const uint4*)(A + (size_t)gr * K + k0 + cc);
      }
      int k  = tid >> 3;
      int ns = (tid & 7) << 3;
      bReg = *(const uint4*)(B + (size_t)(k0 + k) * HDIM + colBase + ns);
    }

    // ---- build fragments from current buffer
    const unsigned short* Asb = As[cur];
    const unsigned short* Bsb = Bs[cur];
    const int ka = hi << 3;                   // 0 | 8  (ISA 16-bit A layout)
    const int kb = hi << 3;                   // 0 | 8  (B lane-half K offset)

    v16bf afr[2];
#pragma unroll
    for (int mt = 0; mt < 2; ++mt) {
      int m = (wave << 5) + (mt << 4) + l15;
      v8bf alo = *(const v8bf*)(&Asb[m * LDA + ka]);
      v8bf ahi = *(const v8bf*)(&Asb[m * LDA + ka + 16]);
      afr[mt] = __builtin_shufflevector(alo, ahi,
          0, 1, 2, 3, 4, 5, 6, 7, 8, 9, 10, 11, 12, 13, 14, 15);
    }
    v16bf bfr[4];
#pragma unroll
    for (int t = 0; t < 4; ++t) {
      int n = (t << 4) + l15;
      v8bf blo = *(const v8bf*)(&Bsb[n * LDA + kb]);
      v8bf bhi = *(const v8bf*)(&Bsb[n * LDA + kb + 16]);
      bfr[t] = __builtin_shufflevector(blo, bhi,
          0, 1, 2, 3, 4, 5, 6, 7, 8, 9, 10, 11, 12, 13, 14, 15);
    }

    // ---- 8 back-to-back WMMAs (independent accumulators)
#pragma unroll
    for (int mt = 0; mt < 2; ++mt)
#pragma unroll
      for (int t = 0; t < 4; ++t)
        acc[mt][t] = __builtin_amdgcn_wmma_f32_16x16x32_bf16(
            false, afr[mt], false, bfr[t], (short)0, acc[mt][t], false, false);

    // ---- stage next tile into the other buffer
    if (more) {
      unsigned short* Asn = As[cur ^ 1];
      unsigned short* Bsn = Bs[cur ^ 1];
#pragma unroll
      for (int i = 0; i < 4; ++i) {
        int chunk = tid + (i << 8);
        int r  = chunk >> 2;
        int cc = (chunk & 3) << 3;
        *(uint4*)(&Asn[r * LDA + cc]) = aReg[i];
      }
      int k  = tid >> 3;
      int ns = (tid & 7) << 3;
      const unsigned short* pv = (const unsigned short*)&bReg;
#pragma unroll
      for (int j = 0; j < 8; ++j) Bsn[(ns + j) * LDA + k] = pv[j];
    }
    __syncthreads();
  }

  // ---- epilogue: C layout (lane<16 -> rows v, lane>=16 -> rows v+8)
#pragma unroll
  for (int t = 0; t < 4; ++t) {
    int n = colBase + (t << 4) + l15;
    float bv = bias[n];
#pragma unroll
    for (int mt = 0; mt < 2; ++mt) {
#pragma unroll
      for (int v = 0; v < 8; ++v) {
        int gr = rowBase + (wave << 5) + (mt << 4) + v + (hi << 3);
        if (gr < M) C[(size_t)gr * HDIM + n] = acc[mt][t][v] + bv;
      }
    }
  }
}

// ------------------------------------------------------------- BN statistics
__global__ void k_bn_stats(const float* __restrict__ Z,
                           float* __restrict__ stats, int N) {
  int f = threadIdx.x;                        // 256 threads = 256 columns
  float s = 0.0f, sq = 0.0f;
  for (int r = blockIdx.x; r < N; r += gridDim.x) {
    float v = Z[(size_t)r * HDIM + f];
    s += v;
    sq += v * v;
  }
  atomicAdd(&stats[f], s);
  atomicAdd(&stats[HDIM + f], sq);
}

// scale = g * rsqrt(var+eps); shift = b - mean*scale   (biased var, jnp.var)
__global__ void k_bn_finalize(const float* __restrict__ stats,
                              const float* __restrict__ g,
                              const float* __restrict__ b,
                              float* __restrict__ sc_sh, float invN) {
  int f = threadIdx.x;
  float mean = stats[f] * invN;
  float var  = stats[HDIM + f] * invN - mean * mean;
  float sc = g[f] * rsqrtf(var + BN_EPS);
  sc_sh[f] = sc;
  sc_sh[HDIM + f] = b[f] - mean * sc;
}

// BN apply + ReLU -> bf16 (feeds the second GEMM)
__global__ void k_bn_apply_bf(const float* __restrict__ Z,
                              const float* __restrict__ sc_sh,
                              unsigned short* __restrict__ outbf, int n) {
  int i = blockIdx.x * blockDim.x + threadIdx.x;
  int stride = gridDim.x * blockDim.x;
  for (; i < n; i += stride) {
    int f = i & (HDIM - 1);
    float v = Z[i] * sc_sh[f] + sc_sh[HDIM + f];
    v = fmaxf(v, 0.0f);
    outbf[i] = f2bf(v);
  }
}

// BN apply (+optional ReLU) -> f32 (layer output / final output)
__global__ void k_bn_apply_f32(const float* __restrict__ Z,
                               const float* __restrict__ sc_sh,
                               float* __restrict__ out, int n, int do_relu) {
  int i = blockIdx.x * blockDim.x + threadIdx.x;
  int stride = gridDim.x * blockDim.x;
  for (; i < n; i += stride) {
    int f = i & (HDIM - 1);
    float v = Z[i] * sc_sh[f] + sc_sh[HDIM + f];
    if (do_relu) v = fmaxf(v, 0.0f);
    out[i] = v;
  }
}

// ---------------------------------------------------------------------------
extern "C" void kernel_launch(void* const* d_in, const int* in_sizes, int n_in,
                              void* d_out, int out_size, void* d_ws, size_t ws_size,
                              hipStream_t stream) {
  const float* x       = (const float*)d_in[0];
  const int*   eidx    = (const int*)d_in[1];
  const float* eps     = (const float*)d_in[2];
  const float* W1_0    = (const float*)d_in[3];
  const float* W1_rest = (const float*)d_in[4];
  const float* b1      = (const float*)d_in[5];
  const float* g1      = (const float*)d_in[6];
  const float* bt1     = (const float*)d_in[7];
  const float* W2      = (const float*)d_in[8];
  const float* b2      = (const float*)d_in[9];
  const float* g2      = (const float*)d_in[10];
  const float* bt2     = (const float*)d_in[11];

  const int D_IN = 128;
  const int N = in_sizes[0] / D_IN;        // 50000
  const int E = in_sizes[1] / 2;           // 800000
  const int L = in_sizes[2];               // 3

  const int* rows = eidx;                  // edge_index[0]
  const int* cols = eidx + E;              // edge_index[1]

  // ---- workspace carve (~180 MB)
  char* ws = (char*)d_ws;
  float* h_buf = (float*)ws;                    ws += (size_t)N * HDIM * sizeof(float);
  float* agg   = (float*)ws;                    ws += (size_t)N * HDIM * sizeof(float);
  float* Z     = (float*)ws;                    ws += (size_t)N * HDIM * sizeof(float);
  unsigned short* A_bf = (unsigned short*)ws;   ws += (size_t)N * HDIM * sizeof(unsigned short);
  unsigned short* W_bf = (unsigned short*)ws;   ws += (size_t)HDIM * HDIM * sizeof(unsigned short);
  float* stats = (float*)ws;                    ws += 2 * HDIM * sizeof(float);
  float* sc_sh = (float*)ws;

  float* out = (float*)d_out;
  const float invN = 1.0f / (float)N;

  for (int i = 0; i < L; ++i) {
    const int D      = (i == 0) ? D_IN : HDIM;
    const int logD4  = (i == 0) ? 5 : 6;       // log2(D/4)
    const float* hsrc = (i == 0) ? x : h_buf;
    const float* Wa   = (i == 0) ? W1_0 : (W1_rest + (size_t)(i - 1) * HDIM * HDIM);
    const int nHD = N * D;

    // aggregation: agg = segment_sum(h[col], row)
    k_zero<<<4096, 256, 0, stream>>>(agg, nHD);
    int aggThreads = E << logD4;
    k_aggregate<<<(aggThreads + 255) / 256, 256, 0, stream>>>(
        hsrc, rows, cols, agg, E, D, logD4);

    // A = bf16((1+eps)*h + agg)
    k_combine<<<4096, 256, 0, stream>>>(hsrc, agg, eps, i, A_bf, nHD);

    // GEMM1: Z = A @ W1 + b1
    k_f2bf<<<256, 256, 0, stream>>>(Wa, W_bf, D * HDIM);
    dim3 gemmGrid((N + 255) / 256, 4);
    k_gemm<<<gemmGrid, 256, 0, stream>>>(A_bf, W_bf, b1 + i * HDIM, Z, N, D);

    // BN1 -> ReLU -> bf16
    k_zero<<<2, 256, 0, stream>>>(stats, 2 * HDIM);
    k_bn_stats<<<512, 256, 0, stream>>>(Z, stats, N);
    k_bn_finalize<<<1, 256, 0, stream>>>(stats, g1 + i * HDIM, bt1 + i * HDIM, sc_sh, invN);
    k_bn_apply_bf<<<4096, 256, 0, stream>>>(Z, sc_sh, A_bf, N * HDIM);

    // GEMM2: Z = A @ W2 + b2
    k_f2bf<<<256, 256, 0, stream>>>(W2 + (size_t)i * HDIM * HDIM, W_bf, HDIM * HDIM);
    k_gemm<<<gemmGrid, 256, 0, stream>>>(A_bf, W_bf, b2 + i * HDIM, Z, N, HDIM);

    // BN2 (+ inter-layer ReLU); last layer writes d_out
    k_zero<<<2, 256, 0, stream>>>(stats, 2 * HDIM);
    k_bn_stats<<<512, 256, 0, stream>>>(Z, stats, N);
    k_bn_finalize<<<1, 256, 0, stream>>>(stats, g2 + i * HDIM, bt2 + i * HDIM, sc_sh, invN);
    float* dst = (i == L - 1) ? out : h_buf;
    k_bn_apply_f32<<<4096, 256, 0, stream>>>(Z, sc_sh, dst, N * HDIM, (i < L - 1) ? 1 : 0);
  }
}
